// NinjaTurtleLinear_47304769798211
// MI455X (gfx1250) — matrix operand order repaired and checked by reading, not compiled, not run on hardware
//
#include <hip/hip_runtime.h>
#include <hip/hip_bf16.h>

typedef _Float16 v8h  __attribute__((ext_vector_type(8)));
typedef _Float16 v16h __attribute__((ext_vector_type(16)));
typedef float    v8f  __attribute__((ext_vector_type(8)));

#define K_DIM 2048
#define N_DIM 8192
#define BM 256
#define BN 256
#define BK 64
#define LDST 72   // 64 + 8 halves pad: 144B row stride, 16B aligned, bank-conflict free

static __device__ __forceinline__ v8h pack8(float4 a, float4 b) {
    v8h h;
    h[0] = (_Float16)a.x; h[1] = (_Float16)a.y; h[2] = (_Float16)a.z; h[3] = (_Float16)a.w;
    h[4] = (_Float16)b.x; h[5] = (_Float16)b.y; h[6] = (_Float16)b.z; h[7] = (_Float16)b.w;
    return h;
}

__global__ __launch_bounds__(512)
void ninja_sparse_linear_wmma_f16(const float* __restrict__ x,
                                  const float* __restrict__ weight,
                                  const float* __restrict__ bias,
                                  const float* __restrict__ mask,
                                  float* __restrict__ out)
{
    __shared__ _Float16 As[2][BM * LDST];
    __shared__ _Float16 Bs[2][BN * LDST];

    const int tid    = threadIdx.x;
    const int lane   = tid & 31;
    const int wid    = tid >> 5;        // 16 waves
    const int laneLo = lane & 15;
    const int laneHi = lane >> 4;
    const int wm     = wid & 3;         // 4 waves in M, 64 rows each
    const int wn     = wid >> 2;        // 4 waves in N, 64 cols each
    const int bm     = blockIdx.y * BM;
    const int bn     = blockIdx.x * BN;

    // fill pattern: 8-float chunk per thread, 64 rows/pass, 4 passes
    const int ldr = tid >> 3;           // 0..63
    const int ldc = (tid & 7) * 8;      // chunk col (same index in f32 and f16 units)

    v8f acc[4][4];
#pragma unroll
    for (int i = 0; i < 4; ++i)
#pragma unroll
        for (int j = 0; j < 4; ++j)
            acc[i][j] = (v8f)0.0f;

    // ---- prologue: stage 0, fused f32->f16 convert and mask*W ----
#pragma unroll
    for (int p = 0; p < 4; ++p) {
        int r = ldr + p * 64;
        const float* xr = &x[(size_t)(bm + r) * K_DIM + ldc];
        const float* wr = &weight[(size_t)(bn + r) * K_DIM + ldc];
        const float* mr = &mask[(size_t)(bn + r) * K_DIM + ldc];
        float4 x0 = *(const float4*)xr,       x1 = *(const float4*)(xr + 4);
        float4 w0 = *(const float4*)wr,       w1 = *(const float4*)(wr + 4);
        float4 m0 = *(const float4*)mr,       m1 = *(const float4*)(mr + 4);
        w0.x *= m0.x; w0.y *= m0.y; w0.z *= m0.z; w0.w *= m0.w;
        w1.x *= m1.x; w1.y *= m1.y; w1.z *= m1.z; w1.w *= m1.w;
        *(v8h*)&As[0][r * LDST + ldc] = pack8(x0, x1);
        *(v8h*)&Bs[0][r * LDST + ldc] = pack8(w0, w1);
    }
    __syncthreads();

    const int NS = K_DIM / BK;  // 32 stages
    for (int ks = 0; ks < NS; ++ks) {
        const int cur = ks & 1;
        const int nxt = cur ^ 1;

        // issue next stage's global loads early; converted to f16 staging regs
        v8h xh[4], wh[4];
        if (ks + 1 < NS) {
            const int k0 = (ks + 1) * BK;
#pragma unroll
            for (int p = 0; p < 4; ++p) {
                int r = ldr + p * 64;
                const float* xr = &x[(size_t)(bm + r) * K_DIM + k0 + ldc];
                const float* wr = &weight[(size_t)(bn + r) * K_DIM + k0 + ldc];
                const float* mr = &mask[(size_t)(bn + r) * K_DIM + k0 + ldc];
                float4 x0 = *(const float4*)xr, x1 = *(const float4*)(xr + 4);
                float4 w0 = *(const float4*)wr, w1 = *(const float4*)(wr + 4);
                float4 m0 = *(const float4*)mr, m1 = *(const float4*)(mr + 4);
                w0.x *= m0.x; w0.y *= m0.y; w0.z *= m0.z; w0.w *= m0.w;
                w1.x *= m1.x; w1.y *= m1.y; w1.z *= m1.z; w1.w *= m1.w;
                xh[p] = pack8(x0, x1);
                wh[p] = pack8(w0, w1);
            }
        }

        const _Float16* Ab = &As[cur][0];
        const _Float16* Bb = &Bs[cur][0];
#pragma unroll
        for (int kk = 0; kk < BK; kk += 32) {
            v16h af[4], bf[4];
#pragma unroll
            for (int i = 0; i < 4; ++i) {
                const _Float16* p = &Ab[(wm * 64 + i * 16 + laneLo) * LDST + kk + 16 * laneHi];
                v8h lo = *(const v8h*)p;
                v8h hi = *(const v8h*)(p + 8);    // contiguous 16 K values
                af[i] = __builtin_shufflevector(lo, hi, 0,1,2,3,4,5,6,7,8,9,10,11,12,13,14,15);
            }
#pragma unroll
            for (int j = 0; j < 4; ++j) {
                const _Float16* p = &Bb[(wn * 64 + j * 16 + laneLo) * LDST + kk + 16 * laneHi];
                v8h lo = *(const v8h*)p;
                v8h hi = *(const v8h*)(p + 8);    // contiguous 16 K values
                bf[j] = __builtin_shufflevector(lo, hi, 0,1,2,3,4,5,6,7,8,9,10,11,12,13,14,15);
            }
#pragma unroll
            for (int i = 0; i < 4; ++i)
#pragma unroll
                for (int j = 0; j < 4; ++j)
                    acc[i][j] = __builtin_amdgcn_wmma_f32_16x16x32_f16(
                        false, af[i], false, bf[j],
                        (short)0, acc[i][j], false, false);
        }

        if (ks + 1 < NS) {
#pragma unroll
            for (int p = 0; p < 4; ++p) {
                int r = ldr + p * 64;
                *(v8h*)&As[nxt][r * LDST + ldc] = xh[p];
                *(v8h*)&Bs[nxt][r * LDST + ldc] = wh[p];
            }
        }
        __syncthreads();
    }

    // ---- epilogue: +bias, f32 store ----
#pragma unroll
    for (int i = 0; i < 4; ++i) {
        int m0 = bm + wm * 64 + i * 16 + 8 * laneHi;
#pragma unroll
        for (int j = 0; j < 4; ++j) {
            int n = bn + wn * 64 + j * 16 + laneLo;
            float bv = bias[n];
#pragma unroll
            for (int r = 0; r < 8; ++r)
                out[(size_t)(m0 + r) * N_DIM + n] = acc[i][j][r] + bv;
        }
    }
}

extern "C" void kernel_launch(void* const* d_in, const int* in_sizes, int n_in,
                              void* d_out, int out_size, void* d_ws, size_t ws_size,
                              hipStream_t stream) {
    const float* x      = (const float*)d_in[0];
    const float* weight = (const float*)d_in[1];
    const float* bias   = (const float*)d_in[2];
    const float* mask   = (const float*)d_in[3];
    float* out          = (float*)d_out;

    const int rowsM = in_sizes[0] / K_DIM;        // 16384
    dim3 grid(N_DIM / BN, rowsM / BM);            // (32, 64)
    ninja_sparse_linear_wmma_f16<<<grid, 512, 0, stream>>>(x, weight, bias, mask, out);
}
